// AttentionBlock_54125177864945
// MI455X (gfx1250) — matrix operand order repaired and checked by reading, not compiled, not run on hardware
//
#include <hip/hip_runtime.h>

#define B_   16
#define C_   512
#define HW_  1024
#define NH_  4
#define HD_  128
#define G_   32
#define CPG_ (C_/G_)
#define EPS_ 1e-5f

typedef __attribute__((ext_vector_type(16))) _Float16 v16h;
typedef __attribute__((ext_vector_type(8)))  _Float16 v8h;
typedef __attribute__((ext_vector_type(8)))  float    v8f;
typedef int v4i __attribute__((vector_size(16)));

// ---- CDNA5 async global->LDS copy (ASYNCcnt path), guarded -------------
#if __has_builtin(__builtin_amdgcn_global_load_async_to_lds_b128)
#define HAVE_ASYNC_LDS 1
#endif

#ifdef HAVE_ASYNC_LDS
#if __has_builtin(__builtin_amdgcn_s_wait_asynccnt)
#define ASYNC_WAIT(n) __builtin_amdgcn_s_wait_asynccnt(n)
#else
#define ASYNC_WAIT(n) asm volatile("s_wait_asynccnt %0" ::"i"(n) : "memory")
#endif
#else
#define ASYNC_WAIT(n) ((void)0)
#endif

__device__ __forceinline__ void async_cp16(const _Float16* g, _Float16* l) {
#ifdef HAVE_ASYNC_LDS
  __builtin_amdgcn_global_load_async_to_lds_b128(
      (__attribute__((address_space(1))) v4i*)(void*)g,
      (__attribute__((address_space(3))) v4i*)(void*)l, 0, 0);
#else
  *(v8h*)l = *(const v8h*)g;
#endif
}

// ---- WMMA fragment helpers (wave32, v_wmma_f32_16x16x32_f16) ------------
__device__ __forceinline__ v16h cat8(v8h lo, v8h hi) {
  return __builtin_shufflevector(lo, hi, 0, 1, 2, 3, 4, 5, 6, 7,
                                 8, 9, 10, 11, 12, 13, 14, 15);
}

// A fragment from plain LDS tile [rows][ld] (ld multiple of 8 halves).
__device__ __forceinline__ v16h frag_a(const _Float16* p, int ld, int lane) {
  const int m = lane & 15, h = lane >> 4;
  const _Float16* r = p + m * ld + h * 16;
  return cat8(*(const v8h*)r, *(const v8h*)(r + 8));
}

// B fragment from swizzled LDS tile: chunk (grp*32+lane) holds the lane's
// 16 K-contiguous halves -> 2x ds_load_b128, no packing.
__device__ __forceinline__ v16h frag_b(const _Float16* base, int grp, int lane) {
  const _Float16* r = base + (((grp << 5) + lane) << 4);
  return cat8(*(const v8h*)r, *(const v8h*)(r + 8));
}

// Destination address of an 8-half granule (8 consecutive K, column n)
// inside a swizzled B tile with NJ 16-column groups.
__device__ __forceinline__ _Float16* bdst(_Float16* base, int NJ, int k0, int n) {
  const int grp = (k0 >> 5) * NJ + (n >> 4);
  const int ln = (n & 15) | (((k0 >> 4) & 1) << 4);
  return base + (((grp << 5) + ln) << 4) + (k0 & 8);
}

__device__ __forceinline__ v8f wmma_f16(v16h a, v16h b, v8f c) {
  return __builtin_amdgcn_wmma_f32_16x16x32_f16(false, a, false, b, (short)0, c,
                                                false, false);
}

// ---- shared double-buffered GEMM main loop ------------------------------
// acc[2][2] += A(64x512 rows m0.., row-stride C) x B^T(128x512 rows n0..)
// As: 2*64*40 halves, Bs: 2*4096 halves.
__device__ __forceinline__ void gemm_mainloop(const _Float16* __restrict__ Asrc,
                                              const _Float16* __restrict__ Bsrc,
                                              int m0, int n0, int tid,
                                              _Float16* As, _Float16* Bs,
                                              v8f acc[2][2]) {
  const int wave = tid >> 5, lane = tid & 31;
  const int wm = wave >> 2, wn = wave & 3;
  const int am = tid >> 2, akc = (tid & 3) << 3;

  auto stage = [&](int sel, int kk) {
    async_cp16(&Asrc[(size_t)(m0 + am) * C_ + kk + akc],
               As + sel * (64 * 40) + am * 40 + akc);
#pragma unroll
    for (int it = 0; it < 2; ++it) {
      const int g = tid + it * 256;
      const int n = g >> 2, k0 = (g & 3) << 3;
      async_cp16(&Bsrc[(size_t)(n0 + n) * C_ + kk + k0],
                 bdst(Bs + sel * 4096, 8, k0, n));
    }
  };

  stage(0, 0);
  for (int kk = 0; kk < C_; kk += 32) {
    const int sel = (kk >> 5) & 1;
    if (kk + 32 < C_) {
      stage(sel ^ 1, kk + 32);   // prefetch next tile into other buffer
      ASYNC_WAIT(3);             // tile `kk` (3 older ops/wave) resident
    } else {
      ASYNC_WAIT(0);
    }
    __syncthreads();
    const _Float16* Asl = As + sel * (64 * 40);
    const _Float16* Bsl = Bs + sel * 4096;
    v16h a0 = frag_a(Asl + (wm * 32 + 0) * 40, 40, lane);
    v16h a1 = frag_a(Asl + (wm * 32 + 16) * 40, 40, lane);
    v16h b0 = frag_b(Bsl, wn * 2 + 0, lane);
    v16h b1 = frag_b(Bsl, wn * 2 + 1, lane);
    acc[0][0] = wmma_f16(a0, b0, acc[0][0]);
    acc[0][1] = wmma_f16(a0, b1, acc[0][1]);
    acc[1][0] = wmma_f16(a1, b0, acc[1][0]);
    acc[1][1] = wmma_f16(a1, b1, acc[1][1]);
    __syncthreads();  // all waves done reading before buffer is re-staged
  }
}

// ---- f32 -> f16 conversion ---------------------------------------------
__global__ void cvt_kernel(const float* __restrict__ src,
                           _Float16* __restrict__ dst, int n) {
  int i = blockIdx.x * 256 + threadIdx.x;
  if (i < n) dst[i] = (_Float16)src[i];
}

// ---- GroupNorm -> transposed f16 activations [b][pos][chan] ------------
__global__ void __launch_bounds__(256)
gn_kernel(const float* __restrict__ x, const float* __restrict__ gamma,
          const float* __restrict__ beta, _Float16* __restrict__ hn_t) {
  const int b = blockIdx.x / G_, g = blockIdx.x % G_;
  const int tid = threadIdx.x;
  const size_t base = ((size_t)b * C_ + (size_t)g * CPG_) * HW_;
  const int N = CPG_ * HW_;  // 16384
  float s = 0.f, sq = 0.f;
  for (int i = tid; i < N; i += 256) {  // coalesced stats pass
    float v = x[base + i];
    s += v; sq += v * v;
  }
  __shared__ float ss[256], sv[256];
  __shared__ float s_mean, s_rinv;
  ss[tid] = s; sv[tid] = sq;
  __syncthreads();
  for (int w = 128; w > 0; w >>= 1) {
    if (tid < w) { ss[tid] += ss[tid + w]; sv[tid] += sv[tid + w]; }
    __syncthreads();
  }
  if (tid == 0) {
    float mean = ss[0] / (float)N;
    float var  = sv[0] / (float)N - mean * mean;
    s_mean = mean;
    s_rinv = rsqrtf(var + EPS_);
  }
  __syncthreads();
  const float mean = s_mean, rinv = s_rinv;
  // pass 2: reads hit L2 (group is 64KB), writes coalesced in [pos][chan]
  for (int i = tid; i < N; i += 256) {
    const int cc = i & (CPG_ - 1);
    const int p = i >> 4;
    const int ch = g * CPG_ + cc;
    float v = (x[base + (size_t)cc * HW_ + p] - mean) * rinv * gamma[ch] + beta[ch];
    hn_t[((size_t)b * HW_ + p) * C_ + ch] = (_Float16)v;
  }
}

// ---- Q/K projection: M=pos(1024), N=chan(512), K=chan(512) -------------
__global__ void __launch_bounds__(256)
gemm_qk_kernel(const _Float16* __restrict__ A_t, const _Float16* __restrict__ W,
               const float* __restrict__ bias, _Float16* __restrict__ Out_t,
               float scale) {
  __shared__ __align__(16) _Float16 As[2 * 64 * 40];
  __shared__ __align__(16) _Float16 Bs[2 * 8 * 32 * 16];
  const int m0 = blockIdx.x * 64;    // position tile
  const int n0 = blockIdx.y * 128;   // channel tile
  const _Float16* Ab = A_t + (size_t)blockIdx.z * HW_ * C_;
  _Float16* Ob = Out_t + (size_t)blockIdx.z * HW_ * C_;
  const int tid = threadIdx.x;
  const v8f zero = {};
  v8f acc[2][2];
#pragma unroll
  for (int i = 0; i < 2; ++i)
#pragma unroll
    for (int j = 0; j < 2; ++j) acc[i][j] = zero;

  gemm_mainloop(Ab, W, m0, n0, tid, As, Bs, acc);

  const int wave = tid >> 5, lane = tid & 31;
  const int wm = wave >> 2, wn = wave & 3;
  const int half = lane >> 4, nn = lane & 15;
#pragma unroll
  for (int ti = 0; ti < 2; ++ti)
#pragma unroll
    for (int tj = 0; tj < 2; ++tj)
#pragma unroll
      for (int r = 0; r < 8; ++r) {
        const int m = m0 + wm * 32 + ti * 16 + r + half * 8;  // position
        const int n = n0 + wn * 32 + tj * 16 + nn;            // channel
        Ob[(size_t)m * C_ + n] = (_Float16)((acc[ti][tj][r] + bias[n]) * scale);
      }
}

// ---- V projection: M=chan(512), N=pos(1024), K=chan(512) ---------------
__global__ void __launch_bounds__(256)
gemm_cp_kernel(const _Float16* __restrict__ W, const _Float16* __restrict__ X_t,
               const float* __restrict__ bias, _Float16* __restrict__ Out) {
  __shared__ __align__(16) _Float16 As[2 * 64 * 40];
  __shared__ __align__(16) _Float16 Bs[2 * 8 * 32 * 16];
  const int m0 = blockIdx.x * 64;    // channel tile
  const int n0 = blockIdx.y * 128;   // position tile
  const _Float16* Xb = X_t + (size_t)blockIdx.z * HW_ * C_;
  _Float16* Ob = Out + (size_t)blockIdx.z * C_ * HW_;
  const int tid = threadIdx.x;
  const v8f zero = {};
  v8f acc[2][2];
#pragma unroll
  for (int i = 0; i < 2; ++i)
#pragma unroll
    for (int j = 0; j < 2; ++j) acc[i][j] = zero;

  gemm_mainloop(W, Xb, m0, n0, tid, As, Bs, acc);

  const int wave = tid >> 5, lane = tid & 31;
  const int wm = wave >> 2, wn = wave & 3;
  const int half = lane >> 4, nn = lane & 15;
#pragma unroll
  for (int ti = 0; ti < 2; ++ti)
#pragma unroll
    for (int tj = 0; tj < 2; ++tj)
#pragma unroll
      for (int r = 0; r < 8; ++r) {
        const int m = m0 + wm * 32 + ti * 16 + r + half * 8;  // channel
        const int n = n0 + wn * 32 + tj * 16 + nn;            // position
        Ob[(size_t)m * HW_ + n] = (_Float16)(acc[ti][tj][r] + bias[m]);
      }
}

// ---- Output projection + residual (f32 out) ----------------------------
__global__ void __launch_bounds__(256)
gemm_proj_kernel(const _Float16* __restrict__ W, const _Float16* __restrict__ X_t,
                 const float* __restrict__ bias, const float* __restrict__ resid,
                 float* __restrict__ Out) {
  __shared__ __align__(16) _Float16 As[2 * 64 * 40];
  __shared__ __align__(16) _Float16 Bs[2 * 8 * 32 * 16];
  const int m0 = blockIdx.x * 64;
  const int n0 = blockIdx.y * 128;
  const _Float16* Xb = X_t + (size_t)blockIdx.z * HW_ * C_;
  const size_t boff = (size_t)blockIdx.z * C_ * HW_;
  const int tid = threadIdx.x;
  const v8f zero = {};
  v8f acc[2][2];
#pragma unroll
  for (int i = 0; i < 2; ++i)
#pragma unroll
    for (int j = 0; j < 2; ++j) acc[i][j] = zero;

  gemm_mainloop(W, Xb, m0, n0, tid, As, Bs, acc);

  const int wave = tid >> 5, lane = tid & 31;
  const int wm = wave >> 2, wn = wave & 3;
  const int half = lane >> 4, nn = lane & 15;
#pragma unroll
  for (int ti = 0; ti < 2; ++ti)
#pragma unroll
    for (int tj = 0; tj < 2; ++tj)
#pragma unroll
      for (int r = 0; r < 8; ++r) {
        const int m = m0 + wm * 32 + ti * 16 + r + half * 8;
        const int n = n0 + wn * 32 + tj * 16 + nn;
        const size_t idx = boff + (size_t)m * HW_ + n;
        Out[idx] = resid[idx] + acc[ti][tj][r] + bias[m];
      }
}

// ---- Flash attention ----------------------------------------------------
// q_t,k_t: [b][pos][C] f16 (softmax scale folded into q); v: [b][C][pos].
__global__ void __launch_bounds__(128)
flash_attn_kernel(const _Float16* __restrict__ Qg, const _Float16* __restrict__ Kg,
                  const _Float16* __restrict__ Vg, _Float16* __restrict__ Og) {
  __shared__ __align__(16) _Float16 Qs[64][136];      // [qi][d] plain (A-side)
  __shared__ __align__(16) _Float16 Ks[16 * 32 * 16]; // [K=d:128][N=ki:64] swizzled
  __shared__ __align__(16) _Float16 Vs[16 * 32 * 16]; // [K=ki:64][N=d:128] swizzled
  __shared__ __align__(16) _Float16 Ps[64][72];       // [qi][ki] plain (A-side)
  const int q0 = blockIdx.x * 64;
  const int hh = blockIdx.y, bb = blockIdx.z;
  const size_t rowbase = (size_t)bb * HW_;
  const int coff = hh * HD_;
  const _Float16* Vh = Vg + ((size_t)bb * C_ + coff) * HW_;
  const int tid = threadIdx.x, wave = tid >> 5, lane = tid & 31;
  const int half = lane >> 4, nn = lane & 15;

#pragma unroll
  for (int it = 0; it < 8; ++it) {  // Q tile: async copy, d-contiguous
    const int g = tid + it * 128;
    const int qi = g >> 4, dc = (g & 15) << 3;
    async_cp16(&Qg[(rowbase + q0 + qi) * C_ + coff + dc], &Qs[qi][dc]);
  }

  const v8f zero = {};
  float m_i[8], l_i[8];
  v8f O[8];
#pragma unroll
  for (int r = 0; r < 8; ++r) { m_i[r] = -1e30f; l_i[r] = 0.f; }
#pragma unroll
  for (int j = 0; j < 8; ++j) O[j] = zero;

  for (int kt = 0; kt < HW_ / 64; ++kt) {
    const int k0p = kt * 64;
    __syncthreads();  // previous iteration's LDS consumers done
#pragma unroll
    for (int it = 0; it < 8; ++it) {  // Ks: per-lane d-contiguous
      const int g = tid + it * 128;
      const int ki = g >> 4, k0 = (g & 15) << 3;
      async_cp16(&Kg[(rowbase + k0p + ki) * C_ + coff + k0], bdst(Ks, 4, k0, ki));
    }
#pragma unroll
    for (int it = 0; it < 8; ++it) {  // Vs: per-lane ki-contiguous
      const int g = tid + it * 128;
      const int d = g >> 3, k0 = (g & 7) << 3;
      async_cp16(&Vh[(size_t)d * HW_ + k0p + k0], bdst(Vs, 8, k0, d));
    }
    ASYNC_WAIT(0);
    __syncthreads();

    // S (16 q-rows per wave x 64 k-cols) = Q K^T over K-dim HD=128
    v8f S[4];
#pragma unroll
    for (int j = 0; j < 4; ++j) S[j] = zero;
#pragma unroll
    for (int dk = 0; dk < 4; ++dk) {
      v16h a = frag_a(&Qs[wave * 16][dk * 32], 136, lane);
#pragma unroll
      for (int j = 0; j < 4; ++j) {
        v16h bf = frag_b(Ks, dk * 4 + j, lane);
        S[j] = wmma_f16(a, bf, S[j]);
      }
    }

    // Online softmax (rows live in 16-lane halves of the wave)
    float alpha[8];
#pragma unroll
    for (int r = 0; r < 8; ++r) {
      float mx = -1e30f;
#pragma unroll
      for (int j = 0; j < 4; ++j) mx = fmaxf(mx, S[j][r]);
      mx = fmaxf(mx, __shfl_xor(mx, 1, 32));
      mx = fmaxf(mx, __shfl_xor(mx, 2, 32));
      mx = fmaxf(mx, __shfl_xor(mx, 4, 32));
      mx = fmaxf(mx, __shfl_xor(mx, 8, 32));
      const float mnew = fmaxf(m_i[r], mx);
      alpha[r] = __expf(m_i[r] - mnew);
      float rs = 0.f;
#pragma unroll
      for (int j = 0; j < 4; ++j) {
        const float p = __expf(S[j][r] - mnew);
        S[j][r] = p;
        rs += p;
      }
      rs += __shfl_xor(rs, 1, 32);
      rs += __shfl_xor(rs, 2, 32);
      rs += __shfl_xor(rs, 4, 32);
      rs += __shfl_xor(rs, 8, 32);
      l_i[r] = l_i[r] * alpha[r] + rs;
      m_i[r] = mnew;
    }
#pragma unroll
    for (int j = 0; j < 8; ++j)
#pragma unroll
      for (int r = 0; r < 8; ++r) O[j][r] *= alpha[r];

    // P -> LDS (A-side layout), then O += P V^T over K=64
#pragma unroll
    for (int j = 0; j < 4; ++j)
#pragma unroll
      for (int r = 0; r < 8; ++r)
        Ps[wave * 16 + r + half * 8][j * 16 + nn] = (_Float16)S[j][r];
    __syncthreads();
#pragma unroll
    for (int dk = 0; dk < 2; ++dk) {
      v16h a = frag_a(&Ps[wave * 16][dk * 32], 72, lane);
#pragma unroll
      for (int j = 0; j < 8; ++j) {
        v16h bf = frag_b(Vs, dk * 8 + j, lane);
        O[j] = wmma_f16(a, bf, O[j]);
      }
    }
  }

  // Normalize and store to ao_t [b][pos][chan]: lane-contiguous channels
#pragma unroll
  for (int r = 0; r < 8; ++r) l_i[r] = 1.f / l_i[r];
#pragma unroll
  for (int j = 0; j < 8; ++j)
#pragma unroll
    for (int r = 0; r < 8; ++r) {
      const int qi = q0 + wave * 16 + r + half * 8;
      const int ch = coff + j * 16 + nn;
      Og[(rowbase + qi) * C_ + ch] = (_Float16)(O[j][r] * l_i[r]);
    }
}

// ---- host launch --------------------------------------------------------
extern "C" void kernel_launch(void* const* d_in, const int* in_sizes, int n_in,
                              void* d_out, int out_size, void* d_ws, size_t ws_size,
                              hipStream_t stream) {
  (void)in_sizes; (void)n_in; (void)out_size; (void)ws_size;
  const float* x     = (const float*)d_in[0];
  const float* gamma = (const float*)d_in[1];
  const float* beta  = (const float*)d_in[2];
  const float* wq = (const float*)d_in[3];
  const float* bq = (const float*)d_in[4];
  const float* wk = (const float*)d_in[5];
  const float* bk = (const float*)d_in[6];
  const float* wv = (const float*)d_in[7];
  const float* bv = (const float*)d_in[8];
  const float* wp = (const float*)d_in[9];
  const float* bp = (const float*)d_in[10];
  float* out = (float*)d_out;

  char* ws = (char*)d_ws;
  size_t off = 0;
  auto alloc = [&](size_t bytes) {
    char* p = ws + off;
    off += (bytes + 255) & ~(size_t)255;
    return p;
  };
  const size_t wbytes = (size_t)C_ * C_ * sizeof(_Float16);
  const size_t tbytes = (size_t)B_ * C_ * HW_ * sizeof(_Float16);
  _Float16* wqh = (_Float16*)alloc(wbytes);
  _Float16* wkh = (_Float16*)alloc(wbytes);
  _Float16* wvh = (_Float16*)alloc(wbytes);
  _Float16* wph = (_Float16*)alloc(wbytes);
  _Float16* hnt = (_Float16*)alloc(tbytes);  // [b][pos][chan]
  _Float16* qt  = (_Float16*)alloc(tbytes);  // [b][pos][chan]
  _Float16* kt  = (_Float16*)alloc(tbytes);  // [b][pos][chan]
  _Float16* vh  = (_Float16*)alloc(tbytes);  // [b][chan][pos]
  _Float16* aot = (_Float16*)alloc(tbytes);  // [b][pos][chan]

  const int wn = C_ * C_;
  cvt_kernel<<<(wn + 255) / 256, 256, 0, stream>>>(wq, wqh, wn);
  cvt_kernel<<<(wn + 255) / 256, 256, 0, stream>>>(wk, wkh, wn);
  cvt_kernel<<<(wn + 255) / 256, 256, 0, stream>>>(wv, wvh, wn);
  cvt_kernel<<<(wn + 255) / 256, 256, 0, stream>>>(wp, wph, wn);

  gn_kernel<<<B_ * G_, 256, 0, stream>>>(x, gamma, beta, hnt);

  const float scale = 0.08838834764831845f;  // HD^-0.5, folded into q
  dim3 qkgrid(HW_ / 64, C_ / 128, B_);
  gemm_qk_kernel<<<qkgrid, 256, 0, stream>>>(hnt, wqh, bq, qt, scale);
  gemm_qk_kernel<<<qkgrid, 256, 0, stream>>>(hnt, wkh, bk, kt, 1.0f);

  dim3 cpgrid(C_ / 64, HW_ / 128, B_);
  gemm_cp_kernel<<<cpgrid, 256, 0, stream>>>(wvh, hnt, bv, vh);

  flash_attn_kernel<<<dim3(HW_ / 64, NH_, B_), 128, 0, stream>>>(qt, kt, vh, aot);

  gemm_proj_kernel<<<cpgrid, 256, 0, stream>>>(wph, aot, bp, x, out);
}
